// LinearAttention_47691316855114
// MI455X (gfx1250) — matrix operand order repaired and checked by reading, not compiled, not run on hardware
//
#include <hip/hip_runtime.h>

// ---------------- problem constants ----------------
#define BATCH   8
#define DIM     512          // C
#define HEADS   8
#define DHEAD   64
#define SEQ     4096         // L
#define F3      1536         // 3*H*D
#define QK_SCALE 0.125f      // 64^-0.5

typedef __attribute__((ext_vector_type(16))) __bf16 v16bf;
typedef __attribute__((ext_vector_type(8)))  __bf16 v8bf;
typedef __attribute__((ext_vector_type(8)))  float  v8f;

// ---------------- WMMA helpers (CDNA5 bf16, 16x16x32, fp32 accum) ----------------
__device__ __forceinline__ v8f wmma_bf16(v16bf a, v16bf b, v8f c) {
    // (neg_a, A, neg_b, B, c_mod, C, reuse_a, reuse_b)
    return __builtin_amdgcn_wmma_f32_16x16x32_bf16(false, a, false, b, (short)0, c, false, false);
}

// A fragment 16x32 (MxK), A row-major with leading dim lda.
// lanes 0-15: halfs = K[k..k+7], K[k+16..k+23];  lanes 16-31: +8 offset.
__device__ __forceinline__ v16bf load_a_frag(const __bf16* __restrict__ A, int lda,
                                             int m, int k, int lane) {
    int row = m + (lane & 15);
    int hi  = (lane >> 4) & 1;
    const __bf16* p = A + (long)row * lda + k + hi * 8;
    v8bf lo = *(const v8bf*)p;
    v8bf hh = *(const v8bf*)(p + 16);
    return __builtin_shufflevector(lo, hh, 0,1,2,3,4,5,6,7,8,9,10,11,12,13,14,15);
}

// B fragment 32x16 (KxN) where B[k][n] = S[n*lds + k]  (S holds B transposed, row-major).
// lanes 0-15: halfs = K[k..k+7], K[k+16..k+23] of column n;  lanes 16-31: +8 offset.
__device__ __forceinline__ v16bf load_b_frag(const __bf16* __restrict__ S, int lds,
                                             int n, int k, int lane) {
    int col = n + (lane & 15);
    int hi  = (lane >> 4) & 1;
    const __bf16* p = S + (long)col * lds + k + hi * 8;
    v8bf lo = *(const v8bf*)p;
    v8bf hh = *(const v8bf*)(p + 16);
    return __builtin_shufflevector(lo, hh, 0,1,2,3,4,5,6,7,8,9,10,11,12,13,14,15);
}

// ---------------- transpose + fp32->bf16 convert ----------------
// in: (R x S) fp32 row-major  ->  out: (S x R) bf16 row-major. Batched via strides.
// NT loads: source is streamed exactly once; keep L2 for the bf16 intermediates.
__global__ void k_transpose_conv(const float* __restrict__ in, __bf16* __restrict__ out,
                                 int R, int S, long inBatch, long outBatch) {
    __shared__ float tile[32][33];
    const float* ip = in  + (long)blockIdx.z * inBatch;
    __bf16*      op = out + (long)blockIdx.z * outBatch;
    int sBase = blockIdx.x * 32, rBase = blockIdx.y * 32;
    int tx = threadIdx.x, ty = threadIdx.y;               // 32 x 8
#pragma unroll
    for (int j = 0; j < 32; j += 8) {
        int r = rBase + ty + j, s = sBase + tx;
        tile[ty + j][tx] = (r < R && s < S) ? __builtin_nontemporal_load(&ip[(long)r * S + s]) : 0.f;
    }
    __syncthreads();
#pragma unroll
    for (int j = 0; j < 32; j += 8) {
        int s = sBase + ty + j, r = rBase + tx;
        if (s < S && r < R) op[(long)s * R + r] = (__bf16)tile[tx][ty + j];
    }
}

// ---------------- GEMM1: qkv projection ----------------
// D[f][l] = sum_c wqkvT[f][c] * x[b][c][l], with B-src = xt (L x C) bf16.
// Each wave computes a 64x64 tile (16 WMMA accs): 16 b128 loads : 16 WMMAs per K-step.
// f<512 -> q (scaled, transposed to (B,H,L,64)); 512..1023 -> k logits (B,H,64,L);
// 1024..1535 -> v (B,H,64,L).
__global__ void __launch_bounds__(256, 1)
k_gemm_qkv(const __bf16* __restrict__ Aw,   // wqkvT (1536 x 512)
           const __bf16* __restrict__ Xt,   // (B, L, 512)
           __bf16* __restrict__ qt,         // (B,H,4096,64)
           __bf16* __restrict__ kl,         // (B,H,64,4096)
           __bf16* __restrict__ vb) {       // (B,H,64,4096)
    const int b    = blockIdx.z;
    const int mB   = blockIdx.y * 256;
    const int nB   = blockIdx.x * 128;
    const int w    = threadIdx.x >> 5;
    const int lane = threadIdx.x & 31;
    const int mW   = mB + (w >> 1) * 64;    // wave: 64 rows
    const int nW   = nB + (w & 1) * 64;     // wave: 64 cols
    const __bf16* Bx = Xt + (long)b * SEQ * DIM;

    v8f acc[4][4] = {};
    for (int k = 0; k < DIM; k += 32) {
        v16bf a[4], bb[4];
#pragma unroll
        for (int i = 0; i < 4; ++i) a[i]  = load_a_frag(Aw, DIM, mW + 16 * i, k, lane);
#pragma unroll
        for (int j = 0; j < 4; ++j) bb[j] = load_b_frag(Bx, DIM, nW + 16 * j, k, lane);
        __builtin_prefetch(Aw + (long)(mW + (lane & 15)) * DIM + k + 32, 0, 1);
        __builtin_prefetch(Bx + (long)(nW + (lane & 15)) * DIM + k + 32, 0, 1);
#pragma unroll
        for (int i = 0; i < 4; ++i)
#pragma unroll
            for (int j = 0; j < 4; ++j)
                acc[i][j] = wmma_bf16(a[i], bb[j], acc[i][j]);
    }

    const int hi = (lane >> 4) & 1;
#pragma unroll
    for (int mi = 0; mi < 4; ++mi) {
#pragma unroll
        for (int ni = 0; ni < 4; ++ni) {
            const int mT = mW + mi * 16, nT = nW + ni * 16;
            const int l  = nT + (lane & 15);
            const int m0 = mT + hi * 8;              // first of 8 consecutive f in this lane
            v8f c = acc[mi][ni];
            if (mT < 512) {                          // q: scale + transpose to (B,H,L,64)
                const int h = m0 >> 6, d0 = m0 & 63;
                v8bf r;
#pragma unroll
                for (int i = 0; i < 8; ++i) r[i] = (__bf16)(c[i] * QK_SCALE);
                *(v8bf*)(qt + (((long)(b * HEADS + h) * SEQ + l) << 6) + d0) = r;
            } else if (mT < 1024) {                  // k logits (B,H,64,L)
                const int f2 = m0 - 512, h = f2 >> 6, d0 = f2 & 63;
#pragma unroll
                for (int i = 0; i < 8; ++i)
                    kl[((long)((b * HEADS + h) * DHEAD + d0 + i)) * SEQ + l] = (__bf16)c[i];
            } else {                                 // v (B,H,64,L)
                const int f2 = m0 - 1024, h = f2 >> 6, d0 = f2 & 63;
#pragma unroll
                for (int i = 0; i < 8; ++i)
                    vb[((long)((b * HEADS + h) * DHEAD + d0 + i)) * SEQ + l] = (__bf16)c[i];
            }
        }
    }
}

// ---------------- softmax over L, in-place on bf16 k logits ----------------
__global__ void k_softmax_rows(__bf16* __restrict__ kl) {
    __bf16* p = kl + (long)blockIdx.x * SEQ;
    const int tid = threadIdx.x;
    float vals[16];
    float m = -3.0e38f;
#pragma unroll
    for (int i = 0; i < 16; ++i) { vals[i] = (float)p[tid + i * 256]; m = fmaxf(m, vals[i]); }
    __shared__ float red[256];
    red[tid] = m; __syncthreads();
    for (int s = 128; s > 0; s >>= 1) { if (tid < s) red[tid] = fmaxf(red[tid], red[tid + s]); __syncthreads(); }
    m = red[0]; __syncthreads();
    float sum = 0.f;
#pragma unroll
    for (int i = 0; i < 16; ++i) { vals[i] = __expf(vals[i] - m); sum += vals[i]; }
    red[tid] = sum; __syncthreads();
    for (int s = 128; s > 0; s >>= 1) { if (tid < s) red[tid] += red[tid + s]; __syncthreads(); }
    const float inv = 1.0f / red[0];
#pragma unroll
    for (int i = 0; i < 16; ++i) p[tid + i * 256] = (__bf16)(vals[i] * inv);
}

// ---------------- context: ctx[d][e] = sum_n k_sm[d][n] * v[e][n], store ctxT[e][d] bf16 ----------------
__global__ void k_gemm_ctx(const __bf16* __restrict__ kl, const __bf16* __restrict__ vb,
                           __bf16* __restrict__ ctxT) {
    const int bh   = blockIdx.x;                       // 0..63
    const __bf16* A  = kl + (long)bh * DHEAD * SEQ;    // (64 x 4096) row-major
    const __bf16* Bv = vb + (long)bh * DHEAD * SEQ;    // B[k][e] = Bv[e*4096 + k]
    __bf16* out = ctxT + (long)bh * DHEAD * DHEAD;
    const int w = threadIdx.x >> 5, lane = threadIdx.x & 31;
    const int mW = (w >> 1) * 16;                      // 4 m-tiles
    const int nW = (w & 1) * 32;                       // 2 n-tiles each
    v8f acc[2] = {};
    for (int k = 0; k < SEQ; k += 32) {
        v16bf a  = load_a_frag(A,  SEQ, mW,      k, lane);
        v16bf b0 = load_b_frag(Bv, SEQ, nW,      k, lane);
        v16bf b1 = load_b_frag(Bv, SEQ, nW + 16, k, lane);
        acc[0] = wmma_bf16(a, b0, acc[0]);
        acc[1] = wmma_bf16(a, b1, acc[1]);
    }
    const int hi = (lane >> 4) & 1;
#pragma unroll
    for (int ni = 0; ni < 2; ++ni) {
        const int e  = nW + ni * 16 + (lane & 15);
        const int d0 = mW + hi * 8;
        v8bf r;
#pragma unroll
        for (int i = 0; i < 8; ++i) r[i] = (__bf16)acc[ni][i];
        *(v8bf*)(out + e * DHEAD + d0) = r;            // transposed store: ctxT[e][d]
    }
}

// ---------------- attention out: out[e][l] = sum_d ctxT[e][d] * q[d][l], q via qt (l x d) ----------------
// store transposed: attnT[b][l][h*64 + e]
__global__ void k_gemm_attn(const __bf16* __restrict__ ctxT, const __bf16* __restrict__ qt,
                            __bf16* __restrict__ attnT) {
    const int bh = blockIdx.y;                         // 0..63
    const int b = bh >> 3, h = bh & 7;
    const __bf16* A  = ctxT + (long)bh * DHEAD * DHEAD;   // (64 x 64) row-major
    const __bf16* Bq = qt   + (long)bh * SEQ * DHEAD;     // B[k=d][n=l] = Bq[l*64 + d]
    const int w = threadIdx.x >> 5, lane = threadIdx.x & 31;
    const int nW = blockIdx.x * 128 + w * 16;          // each wave: one 16-wide l tile
    v8f acc[4] = {};
    for (int k = 0; k < DHEAD; k += 32) {
        v16bf bf = load_b_frag(Bq, DHEAD, nW, k, lane);
#pragma unroll
        for (int mt = 0; mt < 4; ++mt) {
            v16bf a = load_a_frag(A, DHEAD, mt * 16, k, lane);
            acc[mt] = wmma_bf16(a, bf, acc[mt]);
        }
    }
    const int hi = (lane >> 4) & 1;
    const int l  = nW + (lane & 15);
#pragma unroll
    for (int mt = 0; mt < 4; ++mt) {
        const int e0 = mt * 16 + hi * 8;
        v8bf r;
#pragma unroll
        for (int i = 0; i < 8; ++i) r[i] = (__bf16)acc[mt][i];
        *(v8bf*)(attnT + ((long)b * SEQ + l) * DIM + h * DHEAD + e0) = r;
    }
}

// ---------------- output projection: out[b][c][l] = sum_f woutT[c][f]*attnT[b][l][f] + bias[c] ----------------
// 64x64 tile per wave; NT stores for the write-once fp32 output.
__global__ void __launch_bounds__(256, 1)
k_gemm_out(const __bf16* __restrict__ Aw,    // woutT (512 x 512)
           const __bf16* __restrict__ At,    // attnT (B, 4096, 512)
           const float* __restrict__ bias,
           float* __restrict__ out) {
    const int b    = blockIdx.z;
    const int mB   = blockIdx.y * 256;
    const int nB   = blockIdx.x * 128;
    const int w    = threadIdx.x >> 5;
    const int lane = threadIdx.x & 31;
    const int mW   = mB + (w >> 1) * 64;
    const int nW   = nB + (w & 1) * 64;
    const __bf16* Bs = At + (long)b * SEQ * DIM;

    v8f acc[4][4] = {};
    for (int k = 0; k < DIM; k += 32) {
        v16bf a[4], bb[4];
#pragma unroll
        for (int i = 0; i < 4; ++i) a[i]  = load_a_frag(Aw, DIM, mW + 16 * i, k, lane);
#pragma unroll
        for (int j = 0; j < 4; ++j) bb[j] = load_b_frag(Bs, DIM, nW + 16 * j, k, lane);
        __builtin_prefetch(Bs + (long)(nW + (lane & 15)) * DIM + k + 32, 0, 1);
#pragma unroll
        for (int i = 0; i < 4; ++i)
#pragma unroll
            for (int j = 0; j < 4; ++j)
                acc[i][j] = wmma_bf16(a[i], bb[j], acc[i][j]);
    }
    const int hi = (lane >> 4) & 1;
#pragma unroll
    for (int mi = 0; mi < 4; ++mi) {
#pragma unroll
        for (int ni = 0; ni < 4; ++ni) {
            const int c0 = mW + mi * 16 + hi * 8;
            const int l  = nW + ni * 16 + (lane & 15);
#pragma unroll
            for (int i = 0; i < 8; ++i)
                __builtin_nontemporal_store(acc[mi][ni][i] + bias[c0 + i],
                                            &out[((long)b * DIM + c0 + i) * SEQ + l]);
        }
    }
}

// ---------------- host launcher ----------------
extern "C" void kernel_launch(void* const* d_in, const int* in_sizes, int n_in,
                              void* d_out, int out_size, void* d_ws, size_t ws_size,
                              hipStream_t stream) {
    (void)in_sizes; (void)n_in; (void)out_size; (void)ws_size;
    const float* x     = (const float*)d_in[0];   // (B, 512, 4096)
    const float* w_qkv = (const float*)d_in[1];   // (512, 1536)
    const float* w_out = (const float*)d_in[2];   // (512, 512)
    const float* b_out = (const float*)d_in[3];   // (512,)
    float* out = (float*)d_out;                   // (B, 512, 4096)

    char* ws = (char*)d_ws;
    size_t off = 0;
    auto take = [&](size_t bytes) -> char* {
        char* p = ws + off;
        off += (bytes + 255) & ~(size_t)255;
        return p;
    };
    __bf16* xt     = (__bf16*)take((size_t)BATCH * SEQ * DIM * 2);           // (B,L,C)
    __bf16* wqkvT  = (__bf16*)take((size_t)F3 * DIM * 2);                    // (1536,512)
    __bf16* woutT  = (__bf16*)take((size_t)DIM * DIM * 2);                   // (512,512)
    __bf16* qt     = (__bf16*)take((size_t)BATCH * HEADS * SEQ * DHEAD * 2); // (B,H,L,64)
    __bf16* kl     = (__bf16*)take((size_t)BATCH * HEADS * DHEAD * SEQ * 2); // (B,H,64,L)
    __bf16* vb     = (__bf16*)take((size_t)BATCH * HEADS * DHEAD * SEQ * 2); // (B,H,64,L)
    __bf16* ctxT   = (__bf16*)take((size_t)BATCH * HEADS * DHEAD * DHEAD * 2);
    __bf16* attnT  = (__bf16*)take((size_t)BATCH * SEQ * DIM * 2);           // (B,L,512)

    const dim3 tb(32, 8);
    // x (B,512,4096) -> xt (B,4096,512)
    k_transpose_conv<<<dim3(SEQ / 32, DIM / 32, BATCH), tb, 0, stream>>>(
        x, xt, DIM, SEQ, (long)DIM * SEQ, (long)SEQ * DIM);
    // w_qkv (512,1536) -> wqkvT (1536,512)
    k_transpose_conv<<<dim3(F3 / 32, DIM / 32, 1), tb, 0, stream>>>(
        w_qkv, wqkvT, DIM, F3, 0, 0);
    // w_out (512,512) -> woutT (512,512)
    k_transpose_conv<<<dim3(DIM / 32, DIM / 32, 1), tb, 0, stream>>>(
        w_out, woutT, DIM, DIM, 0, 0);

    // GEMM1: qkv projection  (M=1536, N=4096, K=512) x B ; 256x128 block tile
    k_gemm_qkv<<<dim3(SEQ / 128, F3 / 256, BATCH), 256, 0, stream>>>(wqkvT, xt, qt, kl, vb);

    // softmax over L on k logits (B*H*D rows)
    k_softmax_rows<<<BATCH * HEADS * DHEAD, 256, 0, stream>>>(kl);

    // context per (b,h): (64 x 64), K=4096
    k_gemm_ctx<<<BATCH * HEADS, 256, 0, stream>>>(kl, vb, ctxT);

    // attention out per (b,h): (64 x 4096), K=64
    k_gemm_attn<<<dim3(SEQ / 128, BATCH * HEADS), 256, 0, stream>>>(ctxT, qt, attnT);

    // output projection (M=512, N=4096, K=512) x B ; 256x128 block tile, NT stores
    k_gemm_out<<<dim3(SEQ / 128, DIM / 256, BATCH), 256, 0, stream>>>(woutT, attnT, b_out, out);
}
